// Attention_39883066311013
// MI455X (gfx1250) — matrix operand order repaired and checked by reading, not compiled, not run on hardware
//
#include <hip/hip_runtime.h>

// GQA flash-attention for gfx1250 (MI455X): f16 WMMA (16x16x32) with f32
// accumulate, online softmax in exp2 domain, distributed row-sum, BK=64 KV
// tiles, double-buffered K/V LDS staging with packed f32->f16 conversion
// (v_cvt_pk_rtz) and vectorized b128/b64 LDS stores.

#define SEQ   2048
#define NH    32
#define NKV   8
#define HD    128
#define QD    (NH * HD)    // 4096
#define KVD   (NKV * HD)   // 1024
#define BQ    128          // q rows per block (8 waves x 16)
#define BK    64           // kv columns per iteration (4 x 16-wide tiles)
#define KPAD  (HD + 8)     // sK row stride (f16), keeps b128 reads conflict-free
#define VPAD  (BK + 8)     // sV/sP row stride (f16)

// softmax in exp2 domain: exp(s*SCALE + mask) == exp2(s*C_S + mask*C_M)
#define C_S   0.1275158164f    // (1/sqrt(128)) / ln2
#define C_M   1.4426950409f    // 1/ln2

typedef __attribute__((ext_vector_type(16))) _Float16 v16h;
typedef __attribute__((ext_vector_type(8)))  _Float16 v8h;
typedef __attribute__((ext_vector_type(4)))  _Float16 v4h;
typedef __attribute__((ext_vector_type(2)))  _Float16 v2h;
typedef __attribute__((ext_vector_type(2)))  __fp16   h2;
typedef __attribute__((ext_vector_type(8)))  float    v8f;
typedef __attribute__((ext_vector_type(4)))  float    v4f;

__device__ __forceinline__ v2h cvt2(float a, float b) {
  h2 t = __builtin_amdgcn_cvt_pkrtz(a, b);
  return __builtin_bit_cast(v2h, t);
}

__device__ __forceinline__ v16h ld16h(const _Float16* p) {
  v8h lo = *(const v8h*)p;
  v8h hi = *(const v8h*)(p + 16);
  return __builtin_shufflevector(lo, hi, 0, 1, 2, 3, 4, 5, 6, 7,
                                         8, 9, 10, 11, 12, 13, 14, 15);
}
__device__ __forceinline__ v8h pk8(v4f x, v4f y) {
  v2h a = cvt2(x[0], x[1]);
  v2h b = cvt2(x[2], x[3]);
  v2h c = cvt2(y[0], y[1]);
  v2h d = cvt2(y[2], y[3]);
  v4h lo = __builtin_shufflevector(a, b, 0, 1, 2, 3);
  v4h hi = __builtin_shufflevector(c, d, 0, 1, 2, 3);
  return __builtin_shufflevector(lo, hi, 0, 1, 2, 3, 4, 5, 6, 7);
}

__global__ __launch_bounds__(256)
void Attention_39883066311013_kernel(const float* __restrict__ q,
                                     const float* __restrict__ k,
                                     const float* __restrict__ v,
                                     const float* __restrict__ mask,
                                     float* __restrict__ out)
{
  __shared__ alignas(16) _Float16 sK[2][BK][KPAD];     // K tile [kv][d]
  __shared__ alignas(16) _Float16 sV[2][HD][VPAD];     // V tile [d][kv] (transposed)
  __shared__ alignas(16) _Float16 sP[8][16][VPAD];     // per-wave P tile [row][kv]

  const int tid  = threadIdx.x;
  const int wave = tid >> 5;
  const int lane = tid & 31;
  const int l16  = lane & 15;
  const int half = lane >> 4;

  const int head = blockIdx.y;
  const int kvh  = head >> 2;             // repeat_interleave(4)
  const int qw   = blockIdx.x * BQ + wave * 16;

  // loader decomposition
  const int klr = tid >> 2;               // K: kv row 0..63
  const int klc = (tid & 3) * 32;         // K: d base
  const int vkv = (tid & 15) * 4;         // V: kv base 0..60
  const int vd  = (tid >> 4) * 8;         // V: d base 0..120

  const float* kgb = k + (size_t)klr * KVD + kvh * HD + klc;
  const float* vgb = v + (size_t)vkv * KVD + kvh * HD + vd;

  // ---- Q strip -> A-fragments (vectorized load + packed cvt) ----
  v16h aQ[4];
  {
    const float* qp = q + (size_t)(qw + l16) * QD + head * HD;
    #pragma unroll
    for (int kk = 0; kk < 4; ++kk) {
      const int d0 = kk * 32 + half * 8;
      v4f a0 = *(const v4f*)(qp + d0);
      v4f a1 = *(const v4f*)(qp + d0 + 4);
      v4f b0 = *(const v4f*)(qp + d0 + 16);
      v4f b1 = *(const v4f*)(qp + d0 + 20);
      v8h lo = pk8(a0, a1);
      v8h hi = pk8(b0, b1);
      aQ[kk] = __builtin_shufflevector(lo, hi, 0, 1, 2, 3, 4, 5, 6, 7,
                                               8, 9, 10, 11, 12, 13, 14, 15);
    }
  }

  // ---- accumulators / softmax state (C layout: row = r + 8*half, col = l16) ----
  v8f o[8];
  {
    v8f z = {};
    #pragma unroll
    for (int j = 0; j < 8; ++j) o[j] = z;
  }
  float mrow[8], lrow[8];   // lrow = lane-distributed partial sums
  #pragma unroll
  for (int r = 0; r < 8; ++r) { mrow[r] = -1.0e30f; lrow[r] = 0.0f; }

  const float* mrowp = mask + (size_t)(qw + half * 8) * SEQ + l16;

  // ---- stage tile 0 ----
  #pragma unroll
  for (int i = 0; i < 4; ++i) {
    const float* kg = kgb + i * 8;
    *(v8h*)&sK[0][klr][klc + i * 8] =
        pk8(*(const v4f*)kg, *(const v4f*)(kg + 4));
  }
  {
    v4f ra[4], rb[4];
    #pragma unroll
    for (int rr = 0; rr < 4; ++rr) {
      const float* vg = vgb + (size_t)rr * KVD;
      ra[rr] = *(const v4f*)vg;
      rb[rr] = *(const v4f*)(vg + 4);
    }
    #pragma unroll
    for (int d = 0; d < 8; ++d) {
      float x0 = (d < 4) ? ra[0][d & 3] : rb[0][d & 3];
      float x1 = (d < 4) ? ra[1][d & 3] : rb[1][d & 3];
      float x2 = (d < 4) ? ra[2][d & 3] : rb[2][d & 3];
      float x3 = (d < 4) ? ra[3][d & 3] : rb[3][d & 3];
      v2h p01 = cvt2(x0, x1);
      v2h p23 = cvt2(x2, x3);
      *(v4h*)&sV[0][vd + d][vkv] =
          __builtin_shufflevector(p01, p23, 0, 1, 2, 3);
    }
  }
  __syncthreads();

  int bsel = 0;
  for (int kvb = 0; kvb < SEQ; kvb += BK) {
    const _Float16 (*cK)[KPAD] = sK[bsel];
    const _Float16 (*cV)[VPAD] = sV[bsel];

    // ---- scores: four 16-wide tiles ----
    v8f s[4];
    #pragma unroll
    for (int t = 0; t < 4; ++t) {
      v16h bK[4];
      #pragma unroll
      for (int kk = 0; kk < 4; ++kk)
        bK[kk] = ld16h(&cK[t * 16 + l16][kk * 32 + half * 8]);
      v8f acc = {};
      #pragma unroll
      for (int kk = 0; kk < 4; ++kk)
        acc = __builtin_amdgcn_wmma_f32_16x16x32_f16(
            false, aQ[kk], false, bK[kk], (short)0, acc, false, false);
      s[t] = acc;
    }

    // ---- scale + additive mask (exp2 domain) ----
    #pragma unroll
    for (int r = 0; r < 8; ++r) {
      const float* mp = mrowp + (size_t)r * SEQ + kvb;
      #pragma unroll
      for (int t = 0; t < 4; ++t)
        s[t][r] = fmaf(s[t][r], C_S, mp[t * 16] * C_M);
    }

    // ---- online softmax: row-max reduction; lrow stays distributed ----
    float alpha[8];
    #pragma unroll
    for (int r = 0; r < 8; ++r) {
      float mx = fmaxf(fmaxf(s[0][r], s[1][r]), fmaxf(s[2][r], s[3][r]));
      #pragma unroll
      for (int off = 8; off > 0; off >>= 1)
        mx = fmaxf(mx, __shfl_xor(mx, off, 32));
      const float mnew = fmaxf(mrow[r], mx);
      alpha[r] = __builtin_amdgcn_exp2f(mrow[r] - mnew);
      mrow[r]  = mnew;
      #pragma unroll
      for (int t = 0; t < 4; ++t)
        s[t][r] = __builtin_amdgcn_exp2f(s[t][r] - mnew);   // p in-place
      lrow[r] = lrow[r] * alpha[r] +
                ((s[0][r] + s[1][r]) + (s[2][r] + s[3][r]));
    }

    #pragma unroll
    for (int j = 0; j < 8; ++j)
      #pragma unroll
      for (int r = 0; r < 8; ++r)
        o[j][r] *= alpha[r];

    // ---- P: C-layout -> per-wave LDS -> two A-fragments ----
    #pragma unroll
    for (int r = 0; r < 8; ++r)
      #pragma unroll
      for (int t = 0; t < 4; ++t)
        sP[wave][half * 8 + r][t * 16 + l16] = (_Float16)s[t][r];

    v16h aP[2];
    {
      const _Float16* pp = &sP[wave][l16][0];
      #pragma unroll
      for (int c = 0; c < 2; ++c) {
        v8h lo = *(const v8h*)(pp + c * 32 + half * 8);
        v8h hi = *(const v8h*)(pp + c * 32 + 16 + half * 8);
        aP[c] = __builtin_shufflevector(lo, hi, 0, 1, 2, 3, 4, 5, 6, 7,
                                                8, 9, 10, 11, 12, 13, 14, 15);
      }
    }

    // ---- O += P @ V ----
    #pragma unroll
    for (int j = 0; j < 8; ++j) {
      v16h b0 = ld16h(&cV[j * 16 + l16][half * 8]);
      v16h b1 = ld16h(&cV[j * 16 + l16][32 + half * 8]);
      o[j] = __builtin_amdgcn_wmma_f32_16x16x32_f16(
          false, aP[0], false, b0, (short)0, o[j], false, false);
      o[j] = __builtin_amdgcn_wmma_f32_16x16x32_f16(
          false, aP[1], false, b1, (short)0, o[j], false, false);
    }

    // ---- stage next tile into the other buffer (uniform branch) ----
    const int nkv = kvb + BK;
    if (nkv < SEQ) {
      _Float16 (*nK)[KPAD] = sK[bsel ^ 1];
      _Float16 (*nV)[VPAD] = sV[bsel ^ 1];
      const float* kg0 = kgb + (size_t)nkv * KVD;
      const float* vg0 = vgb + (size_t)nkv * KVD;
      #pragma unroll
      for (int i = 0; i < 4; ++i) {
        const float* kg = kg0 + i * 8;
        *(v8h*)&nK[klr][klc + i * 8] =
            pk8(*(const v4f*)kg, *(const v4f*)(kg + 4));
      }
      v4f ra[4], rb[4];
      #pragma unroll
      for (int rr = 0; rr < 4; ++rr) {
        const float* vg = vg0 + (size_t)rr * KVD;
        ra[rr] = *(const v4f*)vg;
        rb[rr] = *(const v4f*)(vg + 4);
      }
      #pragma unroll
      for (int d = 0; d < 8; ++d) {
        float x0 = (d < 4) ? ra[0][d & 3] : rb[0][d & 3];
        float x1 = (d < 4) ? ra[1][d & 3] : rb[1][d & 3];
        float x2 = (d < 4) ? ra[2][d & 3] : rb[2][d & 3];
        float x3 = (d < 4) ? ra[3][d & 3] : rb[3][d & 3];
        v2h p01 = cvt2(x0, x1);
        v2h p23 = cvt2(x2, x3);
        *(v4h*)&nV[vd + d][vkv] =
            __builtin_shufflevector(p01, p23, 0, 1, 2, 3);
      }
      if (nkv + BK < SEQ) {
        __builtin_prefetch(kg0 + (size_t)BK * KVD, 0, 1);
        __builtin_prefetch(vg0 + (size_t)BK * KVD, 0, 1);
      }
    }
    __syncthreads();
    bsel ^= 1;
  }

  // ---- final row-sum reduction (once), normalize, store fp32 ----
  #pragma unroll
  for (int r = 0; r < 8; ++r) {
    float ls = lrow[r];
    #pragma unroll
    for (int off = 8; off > 0; off >>= 1)
      ls += __shfl_xor(ls, off, 32);
    const float inv = 1.0f / ls;
    float* op = out + (size_t)(qw + half * 8 + r) * QD + head * HD + l16;
    #pragma unroll
    for (int j = 0; j < 8; ++j)
      op[j * 16] = o[j][r] * inv;
  }
}

extern "C" void kernel_launch(void* const* d_in, const int* in_sizes, int n_in,
                              void* d_out, int out_size, void* d_ws, size_t ws_size,
                              hipStream_t stream) {
  (void)in_sizes; (void)n_in; (void)out_size; (void)d_ws; (void)ws_size;
  const float* q    = (const float*)d_in[0];
  const float* k    = (const float*)d_in[1];
  const float* v    = (const float*)d_in[2];
  const float* mask = (const float*)d_in[3];
  float* out        = (float*)d_out;

  dim3 grid(SEQ / BQ, NH);   // (16, 32)
  dim3 block(256);           // 8 waves (wave32)
  Attention_39883066311013_kernel<<<grid, block, 0, stream>>>(q, k, v, mask, out);
}
